// RoPEAttention_54193897341232
// MI455X (gfx1250) — matrix operand order repaired and checked by reading, not compiled, not run on hardware
//
#include <hip/hip_runtime.h>
#include <hip/hip_bf16.h>
#include <math.h>
#include <stdint.h>

// ---------------------------------------------------------------------------
// Problem constants (fixed by the reference)
// ---------------------------------------------------------------------------
#define BSZ   2
#define NSEQ  2048
#define CDIM  1024
#define NH    16
#define HD    64           // head dim
#define C3    (3 * CDIM)   // 3072
#define MROWS (BSZ * NSEQ) // 4096
#define PITCH 72           // LDS row pitch in bf16 elems (144B: conflict-free, 16B-aligned)

typedef __attribute__((ext_vector_type(16))) __bf16   v16bf;
typedef __attribute__((ext_vector_type(8)))  __bf16   v8bf;
typedef __attribute__((ext_vector_type(8)))  float    v8f;
typedef __attribute__((ext_vector_type(4)))  uint32_t u32x4;
typedef __attribute__((ext_vector_type(4)))  int      i32x4;
typedef __attribute__((ext_vector_type(8)))  int      i32x8;

#if defined(__has_builtin)
#if __has_builtin(__builtin_amdgcn_tensor_load_to_lds)
#define HAVE_TDM 1
#endif
#if __has_builtin(__builtin_amdgcn_update_dpp)
#define HAVE_DPP 1
#endif
#endif

// softmax runs in base 2: scores pre-scaled by (1/sqrt(D))*log2(e), folded into q
// during RoPE. exp2 then maps to the native v_exp_f32.
#define QSCALE (0.125f * 1.4426950408889634f)

__device__ inline float fast_exp2(float x) {
#if defined(__has_builtin) && __has_builtin(__builtin_amdgcn_exp2f)
    return __builtin_amdgcn_exp2f(x);   // raw v_exp_f32 (inputs <= 0; FTZ is fine)
#else
    return exp2f(x);
#endif
}
__device__ inline float fast_rcp(float x) {
#if defined(__has_builtin) && __has_builtin(__builtin_amdgcn_rcpf)
    return __builtin_amdgcn_rcpf(x);
#else
    return 1.0f / x;
#endif
}

// ---- 16-lane butterfly reductions (stay within a DPP row; no cross-half) ----
#ifdef HAVE_DPP
template <int CTRL>
__device__ inline float dpp_movf(float x) {
    return __int_as_float(
        __builtin_amdgcn_update_dpp(0, __float_as_int(x), CTRL, 0xf, 0xf, true));
}
__device__ inline float red16_max(float t) {
    t = fmaxf(t, dpp_movf<0xB1>(t));   // quad_perm(1,0,3,2) : xor 1
    t = fmaxf(t, dpp_movf<0x4E>(t));   // quad_perm(2,3,0,1) : xor 2
    t = fmaxf(t, dpp_movf<0x141>(t));  // row_half_mirror    : xor 7
    t = fmaxf(t, dpp_movf<0x140>(t));  // row_mirror         : xor 15
    return t;
}
__device__ inline float red16_sum(float t) {
    t += dpp_movf<0xB1>(t);
    t += dpp_movf<0x4E>(t);
    t += dpp_movf<0x141>(t);
    t += dpp_movf<0x140>(t);
    return t;
}
#else
__device__ inline float red16_max(float t) {
    t = fmaxf(t, __shfl_xor(t, 1, 32));
    t = fmaxf(t, __shfl_xor(t, 2, 32));
    t = fmaxf(t, __shfl_xor(t, 4, 32));
    t = fmaxf(t, __shfl_xor(t, 8, 32));
    return t;
}
__device__ inline float red16_sum(float t) {
    t += __shfl_xor(t, 1, 32);
    t += __shfl_xor(t, 2, 32);
    t += __shfl_xor(t, 4, 32);
    t += __shfl_xor(t, 8, 32);
    return t;
}
#endif

// D = A*B + C, bf16 inputs, f32 accumulate
__device__ inline v8f wmma_bf16(v16bf a, v16bf b, v8f c) {
    return __builtin_amdgcn_wmma_f32_16x16x32_bf16(
        false, a, false, b, (short)0, c, false, false);
}

// A fragment (16x32): lane holds row = lane%16; runs K = half*8+0..7 and 16+half*8+0..7.
__device__ inline v16bf load_fragA(const __bf16* rowp, int k0, int half) {
    const int kk = k0 + half * 8;
    v8bf lo = *(const v8bf*)(rowp + kk);
    v8bf hi = *(const v8bf*)(rowp + kk + 16);
    v16bf f;
#pragma unroll
    for (int j = 0; j < 8; ++j) { f[j] = lo[j]; f[8 + j] = hi[j]; }
    return f;
}

// B fragment (32x16): lane holds column = lane%16; runs K = half*8+0..7 and 16+half*8+0..7.
__device__ inline v16bf load_fragB(const __bf16* colp, int k0, int half) {
    const int kk = k0 + half * 8;
    v8bf lo = *(const v8bf*)(colp + kk);
    v8bf hi = *(const v8bf*)(colp + kk + 16);
    v16bf f;
#pragma unroll
    for (int j = 0; j < 8; ++j) { f[j] = lo[j]; f[8 + j] = hi[j]; }
    return f;
}

#ifdef HAVE_TDM
// TDM: async-copy a (64 x rows) bf16 tile (row stride = stride_elems) into LDS.
// Rows are 128B; pad 4 DWORDs (16B) after every 32 DWORDs (128B) -> 144B pitch.
__device__ inline void tdm_load_64xR(uint32_t lds_byte_off, const __bf16* gptr,
                                     int rows, int stride_elems) {
    const uint64_t ga = (uint64_t)(uintptr_t)gptr;
    u32x4 g0;
    g0.x = 1u;                                                // count=1, user D#
    g0.y = lds_byte_off;                                      // lds_addr
    g0.z = (uint32_t)ga;                                      // global_addr[31:0]
    g0.w = (uint32_t)((ga >> 32) & 0x1FFFFFFu) | (2u << 30);  // addr[56:32] | type=2
    i32x8 g1;
    g1[0] = (1 << 16)        // data_size = 2 bytes
          | (1 << 20)        // pad_enable
          | (4 << 22)        // pad_interval = 32 DWORDs (128B)
          | (3 << 25);       // pad_amount  =  4 DWORDs (16B) -> 144B pitch
    g1[1] = (64 << 16);      // tensor_dim0 = 64
    g1[2] = (rows << 16);    // tensor_dim1 lo = rows
    g1[3] = (64 << 16);      // tile_dim0 = 64
    g1[4] = rows;            // tile_dim1 = rows
    g1[5] = stride_elems;    // tensor_dim0_stride
    g1[6] = 0;
    g1[7] = 0;
    i32x4 z4 = {0, 0, 0, 0};
#if __has_include(<hip/amd_detail/amd_gfx1250_TDM.h>)
    i32x8 z8 = {0, 0, 0, 0, 0, 0, 0, 0};
    __builtin_amdgcn_tensor_load_to_lds(g0, g1, z4, z4, z8, 0);
#else
    __builtin_amdgcn_tensor_load_to_lds(g0, g1, z4, z4, 0);
#endif
}
#endif

// ---------------------------------------------------------------------------
// Kernel 0: fp32 -> bf16 bulk convert
// ---------------------------------------------------------------------------
__global__ __launch_bounds__(256)
void cvt_f32_bf16(const float* __restrict__ src, __bf16* __restrict__ dst, int n) {
    const int i = (blockIdx.x * 256 + threadIdx.x) * 8;
    if (i + 8 <= n) {
#pragma unroll
        for (int j = 0; j < 8; ++j) dst[i + j] = (__bf16)src[i + j];
    }
}

// ---------------------------------------------------------------------------
// Kernel 1/4: C[m,n] = sum_k A[m,k] * W[n,k] (+bias). bf16 operands, f32 accum.
// Block 256 = 8 waves; block tile 128x128; wave tile 32x64; 64-wide K slabs via TDM.
// ---------------------------------------------------------------------------
template <bool BF16OUT>
__global__ __launch_bounds__(256)
void gemm_wmma_bf16(const __bf16* __restrict__ A, const __bf16* __restrict__ W,
                    const float* __restrict__ bias, void* __restrict__ Cout,
                    int Nd, int K) {
    __shared__ __bf16 asl[128][PITCH];
    __shared__ __bf16 bsl[128][PITCH];

    const int tid  = threadIdx.x;
    const int w    = tid >> 5;
    const int lane = tid & 31;
    const int half = lane >> 4;
    const int l16  = lane & 15;

    const int rowB = blockIdx.y * 128;
    const int colB = blockIdx.x * 128;
    const int wr   = (w & 3) * 32;
    const int wc   = (w >> 2) * 64;

    v8f acc[2][4] = {};

    for (int ks = 0; ks < K; ks += 64) {
        __syncthreads();
#ifdef HAVE_TDM
        if (w == 0) {
            tdm_load_64xR((uint32_t)(uintptr_t)&asl[0][0],
                          A + (size_t)rowB * K + ks, 128, K);
            tdm_load_64xR((uint32_t)(uintptr_t)&bsl[0][0],
                          W + (size_t)colB * K + ks, 128, K);
            __builtin_amdgcn_s_wait_tensorcnt(0);
        }
#else
        {
            const int r = tid >> 1;
            const int c = (tid & 1) * 32;
            const __bf16* ap = A + (size_t)(rowB + r) * K + ks + c;
            const __bf16* bp = W + (size_t)(colB + r) * K + ks + c;
#pragma unroll
            for (int v = 0; v < 4; ++v) {
                *(v8bf*)&asl[r][c + v * 8] = *(const v8bf*)(ap + v * 8);
                *(v8bf*)&bsl[r][c + v * 8] = *(const v8bf*)(bp + v * 8);
            }
        }
#endif
        __syncthreads();

#pragma unroll
        for (int kk = 0; kk < 64; kk += 32) {
            v16bf a0 = load_fragA(&asl[wr + l16][0], kk, half);
            v16bf a1 = load_fragA(&asl[wr + 16 + l16][0], kk, half);
#pragma unroll
            for (int j = 0; j < 4; ++j) {
                v16bf bf = load_fragB(&bsl[wc + j * 16 + l16][0], kk, half);
                acc[0][j] = wmma_bf16(a0, bf, acc[0][j]);
                acc[1][j] = wmma_bf16(a1, bf, acc[1][j]);
            }
        }
    }

#pragma unroll
    for (int i = 0; i < 2; ++i)
#pragma unroll
        for (int j = 0; j < 4; ++j) {
            const int col = colB + wc + j * 16 + l16;
            const float bv = (!BF16OUT && bias) ? bias[col] : 0.0f;
#pragma unroll
            for (int r = 0; r < 8; ++r) {
                const int row = rowB + wr + i * 16 + r + 8 * half;
                if (BF16OUT)
                    ((__bf16*)Cout)[(size_t)row * Nd + col] = (__bf16)acc[i][j][r];
                else
                    ((float*)Cout)[(size_t)row * Nd + col] = acc[i][j][r] + bv;
            }
        }
}

// ---------------------------------------------------------------------------
// Kernel 2: RoPE in-place on bf16 qkv [B,N,3,H,D]; folds score scale into q.
// ---------------------------------------------------------------------------
__global__ __launch_bounds__(256)
void rope_kernel(__bf16* __restrict__ qkv) {
    int i = blockIdx.x * 256 + threadIdx.x; // < B*N*2*H*32
    const int d = i & 31;   i >>= 5;
    const int h = i & 15;   i >>= 4;
    const int s = i & 1;    i >>= 1;
    const int n = i & 2047; i >>= 11;
    const int b = i;

    __bf16* p = qkv + ((size_t)((b * NSEQ + n) * 3 + s) * CDIM) + h * HD + d;

    const float angle = (float)n * __powf(10000.0f, -(float)d * (1.0f / 32.0f));
    float sn, cs;
    __sincosf(angle, &sn, &cs);

    const float qs = (s == 0) ? QSCALE : 1.0f;
    const float v1 = (float)p[0];
    const float v2 = (float)p[32];
    p[0]  = (__bf16)((v1 * cs - v2 * sn) * qs);
    p[32] = (__bf16)((v2 * cs + v1 * sn) * qs);
}

// ---------------------------------------------------------------------------
// Attention tile staging helpers
// ---------------------------------------------------------------------------
__device__ inline void stage_k_tile(__bf16 (*kbuf)[PITCH], const __bf16* kbase,
                                    int kt, int tid) {
#ifdef HAVE_TDM
    if ((tid >> 5) == 0)
        tdm_load_64xR((uint32_t)(uintptr_t)&kbuf[0][0], kbase + (size_t)kt * C3, 64, C3);
#else
    const int srow   = tid >> 2;
    const int schunk = (tid & 3) * 16;
    const __bf16* kp = kbase + (size_t)(kt + srow) * C3 + schunk;
    *(v8bf*)&kbuf[srow][schunk]     = *(const v8bf*)kp;
    *(v8bf*)&kbuf[srow][schunk + 8] = *(const v8bf*)(kp + 8);
#endif
}
__device__ inline void stage_v_tile_t(__bf16 (*vbuf)[PITCH], const __bf16* vbase,
                                      int kt, int tid) {
    const int srow   = tid >> 2;
    const int schunk = (tid & 3) * 16;
    const __bf16* vp = vbase + (size_t)(kt + srow) * C3 + schunk;
    v8bf a = *(const v8bf*)vp;
    v8bf c = *(const v8bf*)(vp + 8);
#pragma unroll
    for (int e = 0; e < 8; ++e) vbuf[schunk + e][srow] = a[e];
#pragma unroll
    for (int e = 0; e < 8; ++e) vbuf[schunk + 8 + e][srow] = c[e];
}

// ---------------------------------------------------------------------------
// Kernel 3: streaming-softmax attention; double-buffered K/V tiles so the TDM
// copy of tile i+1 overlaps the S/softmax/PV compute of tile i.
// ---------------------------------------------------------------------------
__global__ __launch_bounds__(256)
void attn_kernel(const __bf16* __restrict__ qkv, __bf16* __restrict__ attn_out) {
    __shared__ __bf16 klds[2][64][PITCH];    // [buf][key][d]
    __shared__ __bf16 vtlds[2][64][PITCH];   // [buf][d][key] (transposed)
    __shared__ __bf16 plds[8][16][PITCH];    // per-wave P tile

    const int b = blockIdx.z;
    const int h = blockIdx.y;
    const int tid  = threadIdx.x;
    const int w    = tid >> 5;
    const int lane = tid & 31;
    const int half = lane >> 4;
    const int l16  = lane & 15;

    const int qrow0 = blockIdx.x * 128 + w * 16;

    const __bf16* qptr = qkv + ((size_t)(b * NSEQ + qrow0 + l16) * 3 + 0) * CDIM + h * HD;
    v16bf qf0 = load_fragA(qptr, 0, half);
    v16bf qf1 = load_fragA(qptr, 32, half);

    v8f of[4] = {};
    float mrow[8], lrow[8];
#pragma unroll
    for (int r = 0; r < 8; ++r) { mrow[r] = -1e30f; lrow[r] = 0.0f; }

    const __bf16* kbase = qkv + ((size_t)(b * NSEQ) * 3 + 1) * CDIM + h * HD;
    const __bf16* vbase = qkv + ((size_t)(b * NSEQ) * 3 + 2) * CDIM + h * HD;

    // prologue: stage tile 0 into buffer 0
    stage_k_tile(klds[0], kbase, 0, tid);
    stage_v_tile_t(vtlds[0], vbase, 0, tid);

    for (int it = 0; it < NSEQ / 64; ++it) {
        const int cur = it & 1;
#ifdef HAVE_TDM
        if (w == 0) __builtin_amdgcn_s_wait_tensorcnt(0); // K[cur] arrived
#endif
        __syncthreads(); // cur tiles visible; buf[cur^1] free (consumers of it-1 done)

        // kick off next tile's DMA + V transpose while we compute this one
        if (it + 1 < NSEQ / 64) {
            stage_k_tile(klds[cur ^ 1], kbase, (it + 1) * 64, tid);
            stage_v_tile_t(vtlds[cur ^ 1], vbase, (it + 1) * 64, tid);
        }

        // ---- S = Q K^T (16 x 64), scale*log2e already folded into q ----
        v8f sacc[4] = {};
#pragma unroll
        for (int j = 0; j < 4; ++j) {
            const __bf16* kr = &klds[cur][j * 16 + l16][0];
            sacc[j] = wmma_bf16(qf0, load_fragB(kr, 0, half), sacc[j]);
            sacc[j] = wmma_bf16(qf1, load_fragB(kr, 32, half), sacc[j]);
        }

        // ---- online softmax, base 2, DPP butterfly within 16-lane rows ----
#pragma unroll
        for (int r = 0; r < 8; ++r) {
            float t = fmaxf(fmaxf(sacc[0][r], sacc[1][r]), fmaxf(sacc[2][r], sacc[3][r]));
            t = red16_max(t);
            const float mn = fmaxf(mrow[r], t);
            const float alpha = fast_exp2(mrow[r] - mn);
            mrow[r] = mn;
#pragma unroll
            for (int j = 0; j < 4; ++j) sacc[j][r] = fast_exp2(sacc[j][r] - mn);
            float rs = red16_sum(sacc[0][r] + sacc[1][r] + sacc[2][r] + sacc[3][r]);
            lrow[r] = lrow[r] * alpha + rs;
#pragma unroll
            for (int t4 = 0; t4 < 4; ++t4) of[t4][r] *= alpha;
        }

        // ---- P (C-layout) -> LDS bf16 ----
        __bf16* pw = &plds[w][0][0];
#pragma unroll
        for (int j = 0; j < 4; ++j)
#pragma unroll
            for (int r = 0; r < 8; ++r)
                pw[(r + 8 * half) * PITCH + j * 16 + l16] = (__bf16)sacc[j][r];
        __syncthreads(); // P ordered/visible before A-frag reads

        // ---- O += P * V ----
#pragma unroll
        for (int k0i = 0; k0i < 2; ++k0i) {
            v16bf pa = load_fragA(&plds[w][l16][0], k0i * 32, half);
#pragma unroll
            for (int t4 = 0; t4 < 4; ++t4) {
                v16bf vf = load_fragB(&vtlds[cur][t4 * 16 + l16][0], k0i * 32, half);
                of[t4] = wmma_bf16(pa, vf, of[t4]);
            }
        }
    }

    // ---- normalize, write bf16 [B,N,H*D] ----
#pragma unroll
    for (int r = 0; r < 8; ++r) {
        const float inv = fast_rcp(lrow[r]);
        const int qn = qrow0 + r + 8 * half;
#pragma unroll
        for (int t4 = 0; t4 < 4; ++t4) {
            const int col = h * HD + t4 * 16 + l16;
            attn_out[(size_t)(b * NSEQ + qn) * CDIM + col] = (__bf16)(of[t4][r] * inv);
        }
    }
}

// ---------------------------------------------------------------------------
// Launcher
// ---------------------------------------------------------------------------
extern "C" void kernel_launch(void* const* d_in, const int* in_sizes, int n_in,
                              void* d_out, int out_size, void* d_ws, size_t ws_size,
                              hipStream_t stream) {
    const float* x    = (const float*)d_in[0];  // [B,N,C]
    const float* Wqkv = (const float*)d_in[1];  // [3C,C]
    const float* Wout = (const float*)d_in[2];  // [C,C]
    const float* bout = (const float*)d_in[3];  // [C]
    float* out = (float*)d_out;                 // [B,N,C] f32

    // workspace layout (bf16), 48 MB total
    char* ws = (char*)d_ws;
    __bf16* x_bf    = (__bf16*)ws;                     ws += (size_t)MROWS * CDIM * 2;
    __bf16* wqkv_bf = (__bf16*)ws;                     ws += (size_t)C3 * CDIM * 2;
    __bf16* wout_bf = (__bf16*)ws;                     ws += (size_t)CDIM * CDIM * 2;
    __bf16* qkv_bf  = (__bf16*)ws;                     ws += (size_t)MROWS * C3 * 2;
    __bf16* attn_bf = (__bf16*)ws;

    dim3 blk(256);

    cvt_f32_bf16<<<(MROWS * CDIM) / 2048, blk, 0, stream>>>(x, x_bf, MROWS * CDIM);
    cvt_f32_bf16<<<(C3 * CDIM) / 2048, blk, 0, stream>>>(Wqkv, wqkv_bf, C3 * CDIM);
    cvt_f32_bf16<<<(CDIM * CDIM) / 2048, blk, 0, stream>>>(Wout, wout_bf, CDIM * CDIM);

    gemm_wmma_bf16<true><<<dim3(C3 / 128, MROWS / 128), blk, 0, stream>>>(
        x_bf, wqkv_bf, nullptr, qkv_bf, C3, CDIM);

    rope_kernel<<<(BSZ * NSEQ * 2 * NH * 32) / 256, blk, 0, stream>>>(qkv_bf);

    attn_kernel<<<dim3(NSEQ / 128, NH, BSZ), blk, 0, stream>>>(qkv_bf, attn_bf);

    gemm_wmma_bf16<false><<<dim3(CDIM / 128, MROWS / 128), blk, 0, stream>>>(
        attn_bf, wout_bf, bout, out, CDIM, CDIM);
}
